// MultiHeadAttention_40312563040631
// MI455X (gfx1250) — matrix operand order repaired and checked by reading, not compiled, not run on hardware
//
#include <hip/hip_runtime.h>
#include <hip/hip_bf16.h>

#define DM 1024
#define NH 16
#define DH 64
#define SQ 2048

typedef __attribute__((ext_vector_type(16))) __bf16 v16bf;
typedef __attribute__((ext_vector_type(2)))  __bf16 v2bf;
typedef __attribute__((ext_vector_type(16))) unsigned short v16u;
typedef __attribute__((ext_vector_type(8)))  float v8f;
typedef __attribute__((ext_vector_type(4)))  int   v4i;

#if defined(__HIP_DEVICE_COMPILE__) && __has_builtin(__builtin_amdgcn_global_load_async_to_lds_b128)
#define HAVE_ASYNC_LDS 1
#else
#define HAVE_ASYNC_LDS 0
#endif

// native f32 -> bf16 (backend picks v_cvt_pk_bf16_f32 style ops)
__device__ __forceinline__ unsigned short f2bf(float f) {
    return __builtin_bit_cast(unsigned short, (__bf16)f);
}
__device__ __forceinline__ unsigned pkbf(float lo, float hi) {
    v2bf v = { (__bf16)lo, (__bf16)hi };
    return __builtin_bit_cast(unsigned, v);
}

__device__ __forceinline__ v8f wmma_bf16(const v16u& a, const v16u& b, v8f c) {
    return __builtin_amdgcn_wmma_f32_16x16x32_bf16(
        false, __builtin_bit_cast(v16bf, a),
        false, __builtin_bit_cast(v16bf, b),
        (short)0, c, false, false);
}

// 16-byte global -> LDS copy: async (gfx1250 ASYNCcnt path) when available.
__device__ __forceinline__ void cp16(void* lds, const void* gp) {
#if HAVE_ASYNC_LDS
    __builtin_amdgcn_global_load_async_to_lds_b128(
        (__attribute__((address_space(1))) v4i*)(unsigned long long)(uintptr_t)gp,
        (__attribute__((address_space(3))) v4i*)(unsigned int)(uintptr_t)lds,
        0, 0);
#else
    *(uint4*)lds = *(const uint4*)gp;
#endif
}

__device__ __forceinline__ void async_fence() {
#if HAVE_ASYNC_LDS
#if __has_builtin(__builtin_amdgcn_s_wait_asynccnt)
    __builtin_amdgcn_s_wait_asynccnt(0);
#else
    asm volatile("s_wait_asynccnt 0" ::: "memory");
#endif
#endif
}

// ---------------------------------------------------------------------------
// Kernel 1: fused Q/K/V projection GEMM (f32 in -> bf16 out, [B,H,S,Dh]).
// Block tile 128x128, 8 waves, each wave 32x64 (2x4 WMMA tiles): 8 WMMA per
// 6 fragment loads per K-step.  LDS tiles in WMMA-fragment order so each lane
// reads its 16-elem bf16 fragment as one contiguous 32B load.
// ---------------------------------------------------------------------------
__global__ __launch_bounds__(256) void mha_proj_kernel(
    const float* __restrict__ Aq, const float* __restrict__ Ak, const float* __restrict__ Av,
    const float* __restrict__ Wq, const float* __restrict__ Wk, const float* __restrict__ Wv,
    const float* __restrict__ bq, const float* __restrict__ bk, const float* __restrict__ bv,
    unsigned short* __restrict__ Qp, unsigned short* __restrict__ Kp, unsigned short* __restrict__ Vp)
{
    const float* A; const float* W; const float* bias; unsigned short* dst;
    if (blockIdx.z == 0)      { A = Aq; W = Wq; bias = bq; dst = Qp; }
    else if (blockIdx.z == 1) { A = Ak; W = Wk; bias = bk; dst = Kp; }
    else                      { A = Av; W = Wv; bias = bv; dst = Vp; }

    __shared__ alignas(32) unsigned short aF[8 * 32 * 16];  // 8 row-subtiles (128 rows)
    __shared__ alignas(32) unsigned short bF[8 * 32 * 16];  // 8 col-subtiles (128 cols)

    const int t    = threadIdx.x;
    const int lane = t & 31;
    const int w    = t >> 5;
    const int wr   = w & 3;          // wave row: 32 rows
    const int wc   = w >> 2;         // wave col: 64 cols
    const int hf   = (lane >> 4) & 1;
    const int ln   = lane & 15;

    const int mBase = blockIdx.y * 128;
    const int nBase = blockIdx.x * 128;

    // loader mapping
    const int arow = t & 127;            // A: row in tile
    const int agb  = (t >> 7) * 2;       // A: first k-group of 8
    const int wcol = t & 127;            // W: col in tile
    const int wgb  = (t >> 7) * 2;       // W: first k-group of 8

    v8f acc[2][4];
    #pragma unroll
    for (int j = 0; j < 4; ++j) {
        float b0 = bias[nBase + wc * 64 + j * 16 + ln];
        #pragma unroll
        for (int i = 0; i < 2; ++i)
            #pragma unroll
            for (int r = 0; r < 8; ++r) acc[i][j][r] = b0;
    }

    for (int kk = 0; kk < DM; kk += 32) {
        // A tile: 128x32 f32 -> bf16, packed fragment-order 16B stores
        #pragma unroll
        for (int p = 0; p < 2; ++p) {
            int g = agb + p;                                       // 0..3
            const float* src = A + (size_t)(mBase + arow) * DM + kk + g * 8;
            float4 f0 = *(const float4*)(src);
            float4 f1 = *(const float4*)(src + 4);
            uint4 pk;
            pk.x = pkbf(f0.x, f0.y);
            pk.y = pkbf(f0.z, f0.w);
            pk.z = pkbf(f1.x, f1.y);
            pk.w = pkbf(f1.z, f1.w);
            int dl  = (arow & 15) + ((g & 1) << 4);
            int idx = (g >> 1) * 8;
            *(uint4*)&aF[(((arow >> 4) * 32 + dl) << 4) + idx] = pk;
        }
        // W tile: 32x128, column-wise strided loads, packed 16B stores
        #pragma unroll
        for (int p = 0; p < 2; ++p) {
            int g = wgb + p;                                       // 0..3
            const float* wp = W + (size_t)(kk + g * 8) * DM + nBase + wcol;
            uint4 pk;
            pk.x = pkbf(wp[0 * DM], wp[1 * DM]);
            pk.y = pkbf(wp[2 * DM], wp[3 * DM]);
            pk.z = pkbf(wp[4 * DM], wp[5 * DM]);
            pk.w = pkbf(wp[6 * DM], wp[7 * DM]);
            int dl  = (wcol & 15) + ((g & 1) << 4);
            int idx = (g >> 1) * 8;
            *(uint4*)&bF[(((wcol >> 4) * 32 + dl) << 4) + idx] = pk;
        }
        if (kk + 32 < DM)
            __builtin_prefetch(A + (size_t)(mBase + arow) * DM + kk + 32 + agb * 8, 0, 1);
        __syncthreads();

        v16u afr[2], bfr[4];
        #pragma unroll
        for (int i = 0; i < 2; ++i)
            afr[i] = *(const v16u*)&aF[((wr * 2 + i) * 32 + lane) << 4];
        #pragma unroll
        for (int j = 0; j < 4; ++j)
            bfr[j] = *(const v16u*)&bF[((wc * 4 + j) * 32 + lane) << 4];
        #pragma unroll
        for (int i = 0; i < 2; ++i)
            #pragma unroll
            for (int j = 0; j < 4; ++j)
                acc[i][j] = wmma_bf16(afr[i], bfr[j], acc[i][j]);
        __syncthreads();
    }

    // scatter bf16 into [B, H, S, Dh]
    #pragma unroll
    for (int i = 0; i < 2; ++i)
        #pragma unroll
        for (int r = 0; r < 8; ++r) {
            int gm = mBase + wr * 32 + i * 16 + r + (hf ? 8 : 0);
            int b  = gm >> 11, s = gm & 2047;
            #pragma unroll
            for (int j = 0; j < 4; ++j) {
                int gc = nBase + wc * 64 + j * 16 + ln;
                int h  = gc >> 6, d = gc & 63;
                dst[(((size_t)(b * NH + h)) * SQ + s) * DH + d] = f2bf(acc[i][j][r]);
            }
        }
}

// ---------------------------------------------------------------------------
// Kernel 2: flash attention.  K tile row-major (QK^T B-fragments are already
// contiguous 32B reads), V tile stored transposed so P.V B-fragments are
// contiguous too.  K tile streamed with async global->LDS copies.
// ---------------------------------------------------------------------------
__global__ __launch_bounds__(128) void mha_attn_kernel(
    const unsigned short* __restrict__ Qp, const unsigned short* __restrict__ Kp,
    const unsigned short* __restrict__ Vp, unsigned short* __restrict__ Ob)
{
    __shared__ alignas(32) unsigned short lK [64][80];   // [key][dh]
    __shared__ alignas(32) unsigned short lVt[64][80];   // [dh][key] (transposed)
    __shared__ alignas(16) unsigned short lP [4][16][72];

    const int t    = threadIdx.x;
    const int lane = t & 31;
    const int w    = t >> 5;
    const int hf   = (lane >> 4) & 1;
    const int ln   = lane & 15;
    const int bh   = blockIdx.y;
    const int qBase = blockIdx.x * 64;
    const size_t headBase = (size_t)bh * SQ * DH;

    const int krow = t >> 1;             // K: 0..63
    const int kcg  = (t & 1) * 32;       // K: dh chunk base
    const int vpr  = t >> 2;             // V: key row-pair 0..31
    const int vgb  = (t & 3) * 2;        // V: first dh-group of 8

    // Q A-fragments in registers
    v16u qa[2];
    {
        const unsigned short* qptr = Qp + headBase + (size_t)(qBase + w * 16 + ln) * DH;
        #pragma unroll
        for (int c = 0; c < 2; ++c)
            #pragma unroll
            for (int v = 0; v < 8; ++v) {
                int k = c * 32 + hf * 16 + 2 * v;
                qa[c][2 * v]     = qptr[k];
                qa[c][2 * v + 1] = qptr[k + 1];
            }
    }

    float mrow[8], lrow[8];
    v8f Ofr[4];
    #pragma unroll
    for (int r = 0; r < 8; ++r) { mrow[r] = -1e30f; lrow[r] = 0.f; }
    #pragma unroll
    for (int n = 0; n < 4; ++n)
        #pragma unroll
        for (int r = 0; r < 8; ++r) Ofr[n][r] = 0.f;

    for (int kt = 0; kt < SQ; kt += 64) {
        // K tile: async 16B copies, row-major
        {
            const unsigned short* kp = Kp + headBase + (size_t)(kt + krow) * DH + kcg;
            #pragma unroll
            for (int p = 0; p < 4; ++p)
                cp16(&lK[krow][kcg + p * 8], kp + p * 8);
        }
        // V tile: transpose into lVt with u32-packed stores
        #pragma unroll
        for (int p = 0; p < 2; ++p) {
            int dh0 = (vgb + p) * 8;
            const unsigned short* vp0 = Vp + headBase + (size_t)(kt + 2 * vpr) * DH + dh0;
            uint4 a = *(const uint4*)(vp0);
            uint4 b = *(const uint4*)(vp0 + DH);
            const unsigned* aw = &a.x;
            const unsigned* bw = &b.x;
            #pragma unroll
            for (int j = 0; j < 4; ++j) {
                *(unsigned*)&lVt[dh0 + 2 * j][2 * vpr] =
                    (aw[j] & 0xFFFFu) | (bw[j] << 16);
                *(unsigned*)&lVt[dh0 + 2 * j + 1][2 * vpr] =
                    (aw[j] >> 16) | (bw[j] & 0xFFFF0000u);
            }
        }
        if (kt + 64 < SQ) {
            __builtin_prefetch(Kp + headBase + (size_t)(kt + 64 + krow) * DH + kcg, 0, 1);
            __builtin_prefetch(Vp + headBase + (size_t)(kt + 64 + 2 * vpr) * DH + vgb * 8, 0, 1);
        }
        async_fence();
        __syncthreads();

        // S = (Q K^T) * 1/sqrt(Dh); K B-fragments are contiguous 32B reads
        v8f sfr[4];
        #pragma unroll
        for (int n = 0; n < 4; ++n) {
            #pragma unroll
            for (int r = 0; r < 8; ++r) sfr[n][r] = 0.f;
            int key = n * 16 + ln;
            v16u b0 = *(const v16u*)&lK[key][hf * 16];        // dh 0..31
            v16u b1 = *(const v16u*)&lK[key][32 + hf * 16];   // dh 32..63
            sfr[n] = wmma_bf16(qa[0], b0, sfr[n]);
            sfr[n] = wmma_bf16(qa[1], b1, sfr[n]);
            #pragma unroll
            for (int r = 0; r < 8; ++r) sfr[n][r] *= 0.125f;
        }

        // online softmax (reduce across the 16-lane half that holds a row)
        #pragma unroll
        for (int r = 0; r < 8; ++r) {
            float vmax = fmaxf(fmaxf(sfr[0][r], sfr[1][r]), fmaxf(sfr[2][r], sfr[3][r]));
            #pragma unroll
            for (int msk = 1; msk < 16; msk <<= 1)
                vmax = fmaxf(vmax, __shfl_xor(vmax, msk, 32));
            float mnew  = fmaxf(mrow[r], vmax);
            float alpha = __expf(mrow[r] - mnew);
            mrow[r] = mnew;
            lrow[r] *= alpha;
            #pragma unroll
            for (int n = 0; n < 4; ++n) Ofr[n][r] *= alpha;
            float rs = 0.f;
            #pragma unroll
            for (int n = 0; n < 4; ++n) {
                float p = __expf(sfr[n][r] - mnew);
                sfr[n][r] = p;
                rs += p;
            }
            #pragma unroll
            for (int msk = 1; msk < 16; msk <<= 1)
                rs += __shfl_xor(rs, msk, 32);
            lrow[r] += rs;
        }

        // stage P (16 x 64 keys) to per-wave LDS as bf16
        #pragma unroll
        for (int n = 0; n < 4; ++n)
            #pragma unroll
            for (int r = 0; r < 8; ++r)
                lP[w][r + (hf ? 8 : 0)][n * 16 + ln] = f2bf(sfr[n][r]);

        // O += P @ V ; P A-frag = two 16B reads, V B-frag = one 32B read
        #pragma unroll
        for (int c = 0; c < 2; ++c) {
            union { uint4 q[2]; v16u v; } u;
            u.q[0] = *(const uint4*)&lP[w][ln][c * 32 + hf * 16];
            u.q[1] = *(const uint4*)&lP[w][ln][c * 32 + hf * 16 + 8];
            v16u pa = u.v;
            #pragma unroll
            for (int n = 0; n < 4; ++n) {
                int col = n * 16 + ln;
                v16u bv = *(const v16u*)&lVt[col][c * 32 + hf * 16];
                Ofr[n] = wmma_bf16(pa, bv, Ofr[n]);
            }
        }
        __syncthreads();
    }

    // normalize and write bf16 to [B, S, H*Dh]
    const int b = bh >> 4, h = bh & 15;
    #pragma unroll
    for (int r = 0; r < 8; ++r) {
        int srow  = qBase + w * 16 + r + (hf ? 8 : 0);
        float inv = 1.0f / lrow[r];
        #pragma unroll
        for (int n = 0; n < 4; ++n)
            Ob[((size_t)(b * SQ + srow)) * DM + h * DH + n * 16 + ln] =
                f2bf(Ofr[n][r] * inv);
    }
}

// ---------------------------------------------------------------------------
// Kernel 3: output projection.  out f32 = Ob(bf16) @ Wo + bo.
// Same 128x128 tiling; A tile is a pure byte-move into fragment order ->
// async global->LDS copies.
// ---------------------------------------------------------------------------
__global__ __launch_bounds__(256) void mha_out_kernel(
    const unsigned short* __restrict__ A, const float* __restrict__ W,
    const float* __restrict__ bias, float* __restrict__ out)
{
    __shared__ alignas(32) unsigned short aF[8 * 32 * 16];
    __shared__ alignas(32) unsigned short bF[8 * 32 * 16];

    const int t    = threadIdx.x;
    const int lane = t & 31;
    const int w    = t >> 5;
    const int wr   = w & 3;
    const int wc   = w >> 2;
    const int hf   = (lane >> 4) & 1;
    const int ln   = lane & 15;

    const int mBase = blockIdx.y * 128;
    const int nBase = blockIdx.x * 128;

    const int arow = t & 127;
    const int agb  = (t >> 7) * 2;
    const int wcol = t & 127;
    const int wgb  = (t >> 7) * 2;

    v8f acc[2][4];
    #pragma unroll
    for (int j = 0; j < 4; ++j) {
        float b0 = bias[nBase + wc * 64 + j * 16 + ln];
        #pragma unroll
        for (int i = 0; i < 2; ++i)
            #pragma unroll
            for (int r = 0; r < 8; ++r) acc[i][j][r] = b0;
    }

    for (int kk = 0; kk < DM; kk += 32) {
        // A tile: bf16, direct async 16B copies into fragment order
        #pragma unroll
        for (int p = 0; p < 2; ++p) {
            int g = agb + p;
            int dl  = (arow & 15) + ((g & 1) << 4);
            int idx = (g >> 1) * 8;
            cp16(&aF[(((arow >> 4) * 32 + dl) << 4) + idx],
                 A + (size_t)(mBase + arow) * DM + kk + g * 8);
        }
        // W tile: 32x128 f32 -> bf16, column-wise, packed 16B stores
        #pragma unroll
        for (int p = 0; p < 2; ++p) {
            int g = wgb + p;
            const float* wp = W + (size_t)(kk + g * 8) * DM + nBase + wcol;
            uint4 pk;
            pk.x = pkbf(wp[0 * DM], wp[1 * DM]);
            pk.y = pkbf(wp[2 * DM], wp[3 * DM]);
            pk.z = pkbf(wp[4 * DM], wp[5 * DM]);
            pk.w = pkbf(wp[6 * DM], wp[7 * DM]);
            int dl  = (wcol & 15) + ((g & 1) << 4);
            int idx = (g >> 1) * 8;
            *(uint4*)&bF[(((wcol >> 4) * 32 + dl) << 4) + idx] = pk;
        }
        if (kk + 32 < DM)
            __builtin_prefetch(A + (size_t)(mBase + arow) * DM + kk + 32 + agb * 8, 0, 1);
        async_fence();
        __syncthreads();

        v16u afr[2], bfr[4];
        #pragma unroll
        for (int i = 0; i < 2; ++i)
            afr[i] = *(const v16u*)&aF[((wr * 2 + i) * 32 + lane) << 4];
        #pragma unroll
        for (int j = 0; j < 4; ++j)
            bfr[j] = *(const v16u*)&bF[((wc * 4 + j) * 32 + lane) << 4];
        #pragma unroll
        for (int i = 0; i < 2; ++i)
            #pragma unroll
            for (int j = 0; j < 4; ++j)
                acc[i][j] = wmma_bf16(afr[i], bfr[j], acc[i][j]);
        __syncthreads();
    }

    #pragma unroll
    for (int i = 0; i < 2; ++i)
        #pragma unroll
        for (int r = 0; r < 8; ++r) {
            int gm = mBase + wr * 32 + i * 16 + r + (hf ? 8 : 0);
            #pragma unroll
            for (int j = 0; j < 4; ++j) {
                int gc = nBase + wc * 64 + j * 16 + ln;
                out[(size_t)gm * DM + gc] = acc[i][j][r];
            }
        }
}

// ---------------------------------------------------------------------------
extern "C" void kernel_launch(void* const* d_in, const int* in_sizes, int n_in,
                              void* d_out, int out_size, void* d_ws, size_t ws_size,
                              hipStream_t stream) {
    const float* q  = (const float*)d_in[0];
    const float* k  = (const float*)d_in[1];
    const float* v  = (const float*)d_in[2];
    const float* Wq = (const float*)d_in[3];
    const float* bq = (const float*)d_in[4];
    const float* Wk = (const float*)d_in[5];
    const float* bk = (const float*)d_in[6];
    const float* Wv = (const float*)d_in[7];
    const float* bv = (const float*)d_in[8];
    const float* Wo = (const float*)d_in[9];
    const float* bo = (const float*)d_in[10];
    float* out = (float*)d_out;

    const size_t MAT = (size_t)2 * SQ * DM;
    unsigned short* Qp = (unsigned short*)d_ws;
    unsigned short* Kp = Qp + MAT;
    unsigned short* Vp = Kp + MAT;
    unsigned short* Ob = Vp + MAT;

    dim3 gProj(DM / 128, (2 * SQ) / 128, 3);
    mha_proj_kernel<<<gProj, 256, 0, stream>>>(q, k, v, Wq, Wk, Wv, bq, bk, bv,
                                               Qp, Kp, Vp);
    dim3 gAttn(SQ / 64, 2 * NH);
    mha_attn_kernel<<<gAttn, 128, 0, stream>>>(Qp, Kp, Vp, Ob);
    dim3 gOut(DM / 128, (2 * SQ) / 128);
    mha_out_kernel<<<gOut, 256, 0, stream>>>(Ob, Wo, bo, out);
}
